// Hourglass_history_16904991277605
// MI455X (gfx1250) — matrix operand order repaired and checked by reading, not compiled, not run on hardware
//
#include <hip/hip_runtime.h>
#include <hip/hip_bf16.h>
#include <math.h>

typedef __bf16 bf16;
typedef __attribute__((ext_vector_type(16))) __bf16 v16bf;
typedef __attribute__((ext_vector_type(8)))  __bf16 v8bf;
typedef __attribute__((ext_vector_type(8)))  float  v8f;
typedef unsigned int u32x4 __attribute__((ext_vector_type(4)));
typedef int          i32x8 __attribute__((ext_vector_type(8)));
typedef int          i32x4 __attribute__((ext_vector_type(4)));

#define BM 128
#define BN 64
#define BK 64

#if __has_builtin(__builtin_amdgcn_tensor_load_to_lds)
#define GEMM_USE_TDM 1
#else
#define GEMM_USE_TDM 0
#endif
#if !GEMM_USE_TDM && __has_builtin(__builtin_amdgcn_global_load_async_to_lds_b128)
#define GEMM_USE_ASYNC 1
#else
#define GEMM_USE_ASYNC 0
#endif

// ---------------------------------------------------------------------------
// bf16 WMMA GEMM:  C[b] (f32, MxN) (+=) A[b] (bf16 MxK) * B[b] (bf16 KxN)
// Host guarantees: M % 128 == 0, N % 64 == 0, K % 64 == 0.
// convmode==1: B is an activation tensor [Cin, Hd, Wd], K = Cin*9, implicit
// im2col gather for a 3x3/pad-1 convolution (branchless border select).
//
// A tile (128x64, row-major LDS) is loaded by the Tensor Data Mover: one 2D
// descriptor (data_size=2B, tile 64x128, line stride K) issued by wave 0,
// synchronized with s_wait_tensorcnt + workgroup barrier.  Fallbacks:
// per-lane global_load_async_to_lds_b128 (+ s_wait_asynccnt), then plain
// load/store.  B tile is staged in per-lane fragment order (k<->n transpose
// scatter) so each WMMA B operand is one contiguous 32-byte v16bf load.
// BK=64 -> 8 v_wmma_f32_16x16x32_bf16 per barrier interval.
// ---------------------------------------------------------------------------
__global__ __launch_bounds__(256)
void k_wmma_gemm(const bf16* __restrict__ A, long abatch,
                 const bf16* __restrict__ B, long bbatch,
                 float* __restrict__ C, long cbatch,
                 const float* __restrict__ bias,
                 int M, int N, int K,
                 int relu, int accum,
                 int convmode, int Hd, int Wd)
{
    __shared__ __align__(16) bf16 lA[BM * BK];   // row-major [128][64]
    __shared__ __align__(16) bf16 lB[BK * BN];   // fragment-major [4 nsub][2 ks][32 lane][16]

    const int tid  = threadIdx.x;
    const int lane = tid & 31;
    const int wid  = tid >> 5;
    const int wm   = wid & 3;   // 4 waves along M
    const int wn   = wid >> 2;  // 2 waves along N

    A += (long)blockIdx.z * abatch;
    B += (long)blockIdx.z * bbatch;
    C += (long)blockIdx.z * cbatch;

    const int m0 = blockIdx.y * BM;
    const int n0 = blockIdx.x * BN;
    const long HWl = (long)Hd * Wd;

    v8f acc[2][2];
#pragma unroll
    for (int i = 0; i < 2; ++i)
#pragma unroll
        for (int j = 0; j < 2; ++j) { v8f z = {}; acc[i][j] = z; }

    const int rsel = lane >> 4;   // C/D layout: VGPR r holds rows r / r+8
    const int ncol = lane & 15;

    if (accum) {
#pragma unroll
        for (int mt = 0; mt < 2; ++mt)
#pragma unroll
            for (int nt = 0; nt < 2; ++nt)
#pragma unroll
                for (int r = 0; r < 8; ++r)
                    acc[mt][nt][r] = C[(long)(m0 + wm * 32 + mt * 16 + rsel * 8 + r) * N
                                       + n0 + wn * 32 + nt * 16 + ncol];
    }

    for (int k0 = 0; k0 < K; k0 += BK) {
        // ================= stage A tile 128x64 (row-major) =================
#if GEMM_USE_TDM
        if (wid == 0) {
            // 2D tensor descriptor (ISA 8.3/8.4): count=1, type=2 ("image"),
            // data_size=2B, tile_dim0=64 (contig line), tile_dim1=128 rows,
            // tensor_dim0_stride = K elements; dims large => no OOB clipping.
            unsigned lds_base =
                (unsigned)(size_t)(__attribute__((address_space(3))) void*)(void*)lA;
            unsigned long long ga =
                (unsigned long long)(size_t)(A + (long)m0 * K + k0);
            u32x4 g0;
            g0[0] = 1u;                                   // count=1
            g0[1] = lds_base;                             // lds_addr
            g0[2] = (unsigned)ga;                         // global_addr[31:0]
            g0[3] = (unsigned)(ga >> 32) | (2u << 30);    // addr[56:32] | type=2
            const unsigned td0 = 0x7FFFFFFFu, td1 = 0x7FFFFFFFu;
            const unsigned tile0 = BK, tile1 = BM;
            const unsigned long long s0 = (unsigned long long)K;
            i32x8 g1;
            g1[0] = (int)(1u << 16);                      // data_size = 2 bytes
            g1[1] = (int)((td0 & 0xFFFFu) << 16);         // tensor_dim0 lo
            g1[2] = (int)((td0 >> 16) | ((td1 & 0xFFFFu) << 16));
            g1[3] = (int)((td1 >> 16) | (tile0 << 16));   // tile_dim0
            g1[4] = (int)tile1;                           // tile_dim1, tile_dim2=0
            g1[5] = (int)(unsigned)s0;                    // dim0 stride lo
            g1[6] = (int)(unsigned)(s0 >> 32);            // dim0 stride hi
            g1[7] = 0;
            i32x4 gz; gz[0] = gz[1] = gz[2] = gz[3] = 0;
#if __clang_major__ >= 23
            i32x8 gz8;
#pragma unroll
            for (int q = 0; q < 8; ++q) gz8[q] = 0;
            __builtin_amdgcn_tensor_load_to_lds(g0, g1, gz, gz, gz8, 0);
#else
            __builtin_amdgcn_tensor_load_to_lds(g0, g1, gz, gz, 0);
#endif
        }
#elif GEMM_USE_ASYNC
#pragma unroll
        for (int i = 0; i < 4; ++i) {
            int g  = tid + i * 256;        // 0..1023
            int r  = g >> 3;               // 0..127
            int c0 = (g & 7) * 8;          // 0..56
            __builtin_amdgcn_global_load_async_to_lds_b128(
                (__attribute__((address_space(1))) void*)(A + (long)(m0 + r) * K + k0 + c0),
                (__attribute__((address_space(3))) void*)(void*)(lA + r * BK + c0),
                0, 0);
        }
#else
#pragma unroll
        for (int i = 0; i < 4; ++i) {
            int g  = tid + i * 256;
            int r  = g >> 3;
            int c0 = (g & 7) * 8;
            *(v8bf*)(lA + r * BK + c0) = *(const v8bf*)(A + (long)(m0 + r) * K + k0 + c0);
        }
#endif
        // ===== stage B tile 64x64 (k-major per lane: transpose scatter) =====
        if (!convmode) {
#pragma unroll
            for (int i = 0; i < 2; ++i) {
                int g   = tid + i * 256;   // 0..511
                int kk  = g >> 3;          // 0..63
                int n0g = (g & 7) * 8;     // 0..56
                v8bf d = *(const v8bf*)(B + (long)(k0 + kk) * N + n0 + n0g);
                int ks = kk >> 5, cc = kk & 31;
                int khalf = (cc >> 3) & 1, j = (cc & 7) | ((cc >> 4) << 3);
#pragma unroll
                for (int u = 0; u < 8; ++u) {
                    int n    = n0g + u;
                    int nsub = n >> 4;
                    int ln   = khalf * 16 + (n & 15);
                    lB[(nsub * 2 + ks) * 512 + ln * 16 + j] = d[u];
                }
            }
        } else {
#pragma unroll
            for (int i = 0; i < 2; ++i) {
                int g   = tid + i * 256;
                int kk  = g >> 3;
                int n0g = (g & 7) * 8;
                int gk  = k0 + kk;
                int ci  = gk / 9, rem = gk % 9;
                int dy  = rem / 3 - 1, dx = rem % 3 - 1;
                int ks = kk >> 5, cc = kk & 31;
                int khalf = (cc >> 3) & 1, j = (cc & 7) | ((cc >> 4) << 3);
                const bf16* bp = B + (long)ci * HWl;
#pragma unroll
                for (int u = 0; u < 8; ++u) {
                    int n = n0 + n0g + u;
                    int y = n / Wd + dy, x = n % Wd + dx;
                    bool ok = (y >= 0) & (y < Hd) & (x >= 0) & (x < Wd);
                    int yc = min(max(y, 0), Hd - 1);
                    int xc = min(max(x, 0), Wd - 1);
                    bf16 v = bp[(long)yc * Wd + xc];     // always-valid clamped load
                    if (!ok) v = (bf16)0.0f;             // v_cndmask, no exec branch
                    int nn   = n0g + u;
                    int nsub = nn >> 4;
                    int ln   = khalf * 16 + (nn & 15);
                    lB[(nsub * 2 + ks) * 512 + ln * 16 + j] = v;
                }
            }
        }
        // prefetch next B tile toward L2 while this one computes
        if (!convmode && k0 + BK < K)
            __builtin_prefetch(B + (long)(k0 + BK + (tid & 63)) * N + n0, 0, 1);

        // ---- completion of async A stage before the barrier ----
#if GEMM_USE_TDM
#if __has_builtin(__builtin_amdgcn_s_wait_tensorcnt)
        __builtin_amdgcn_s_wait_tensorcnt(0);
#else
        asm volatile("s_wait_tensorcnt 0x0" ::: "memory");
#endif
#elif GEMM_USE_ASYNC
#if __has_builtin(__builtin_amdgcn_s_wait_asynccnt)
        __builtin_amdgcn_s_wait_asynccnt(0);
#else
        asm volatile("s_wait_asynccnt 0x0" ::: "memory");
#endif
#endif
        __syncthreads();

        // ===================== compute 8 WMMA ops =====================
        const v16bf* lBf = (const v16bf*)lB;
        const int rrow = lane & 15, khalf = lane >> 4;
#pragma unroll
        for (int ks = 0; ks < 2; ++ks) {
            v16bf a[2], b[2];
#pragma unroll
            for (int mt = 0; mt < 2; ++mt) {
                int row = wm * 32 + mt * 16 + rrow;
                v8bf lo = *(const v8bf*)(lA + row * BK + ks * 32 + khalf * 8);
                v8bf hi = *(const v8bf*)(lA + row * BK + ks * 32 + 16 + khalf * 8);
                a[mt] = __builtin_shufflevector(lo, hi,
                        0, 1, 2, 3, 4, 5, 6, 7, 8, 9, 10, 11, 12, 13, 14, 15);
            }
#pragma unroll
            for (int nt = 0; nt < 2; ++nt)
                b[nt] = lBf[((wn * 2 + nt) * 2 + ks) * 32 + lane];
            acc[0][0] = __builtin_amdgcn_wmma_f32_16x16x32_bf16(false, a[0], false, b[0], (short)0, acc[0][0], false, false);
            acc[0][1] = __builtin_amdgcn_wmma_f32_16x16x32_bf16(false, a[0], false, b[1], (short)0, acc[0][1], false, false);
            acc[1][0] = __builtin_amdgcn_wmma_f32_16x16x32_bf16(false, a[1], false, b[0], (short)0, acc[1][0], false, false);
            acc[1][1] = __builtin_amdgcn_wmma_f32_16x16x32_bf16(false, a[1], false, b[1], (short)0, acc[1][1], false, false);
        }
        __syncthreads();
    }

    // ---- epilogue: bias / relu / store (shapes are tile-multiples) ----
#pragma unroll
    for (int mt = 0; mt < 2; ++mt)
#pragma unroll
        for (int nt = 0; nt < 2; ++nt)
#pragma unroll
            for (int r = 0; r < 8; ++r) {
                int mm = m0 + wm * 32 + mt * 16 + rsel * 8 + r;
                int nn = n0 + wn * 32 + nt * 16 + ncol;
                float v = acc[mt][nt][r];
                if (bias) v += bias[mm];
                if (relu) v = fmaxf(v, 0.0f);
                C[(long)mm * N + nn] = v;
            }
}

// ---------------------------------------------------------------------------
// f32 -> bf16 (optionally batched with independent strides)
// ---------------------------------------------------------------------------
__global__ void k_cvt_bf16(bf16* __restrict__ dst, long dstride,
                           const float* __restrict__ src, long sstride, long n)
{
    long b = blockIdx.y;
    long i = (long)blockIdx.x * blockDim.x + threadIdx.x;
    if (i < n) dst[b * dstride + i] = (bf16)src[b * sstride + i];
}

__global__ void k_maxpool2(float* __restrict__ out, const float* __restrict__ in,
                           int Ho, int Wo, long total)
{
    long i = (long)blockIdx.x * blockDim.x + threadIdx.x;
    if (i >= total) return;
    int x = (int)(i % Wo);
    long t = i / Wo;
    int y = (int)(t % Ho);
    long bc = t / Ho;
    int Wi = Wo * 2;
    const float* p = in + bc * (long)(Ho * 2) * Wi + (long)(2 * y) * Wi + 2 * x;
    out[i] = fmaxf(fmaxf(p[0], p[1]), fmaxf(p[Wi], p[Wi + 1]));
}

__global__ void k_upadd(float* __restrict__ out, const float* __restrict__ up,
                        const float* __restrict__ low, int Ho, int Wo, long total)
{
    long i = (long)blockIdx.x * blockDim.x + threadIdx.x;
    if (i >= total) return;
    int x = (int)(i % Wo);
    long t = i / Wo;
    int y = (int)(t % Ho);
    long bc = t / Ho;
    int Hl = Ho >> 1, Wl = Wo >> 1;
    out[i] = up[i] + low[bc * (long)Hl * Wl + (long)(y >> 1) * Wl + (x >> 1)];
}

// curr[b, c] for c<D from spa, c>=D from av; mean over HW
__global__ __launch_bounds__(256)
void k_mean2(float* __restrict__ curr, const float* __restrict__ spa,
             const float* __restrict__ av, int D, long sbatch, long abatch, int HW)
{
    int c = blockIdx.x, b = blockIdx.y;
    const float* src = (c < D) ? spa + (long)b * sbatch + (long)c * HW
                               : av + (long)b * abatch + (long)(c - D) * HW;
    __shared__ float red[256];
    float s = 0.f;
    for (int i = threadIdx.x; i < HW; i += 256) s += src[i];
    red[threadIdx.x] = s;
    __syncthreads();
    for (int st = 128; st > 0; st >>= 1) {
        if (threadIdx.x < st) red[threadIdx.x] += red[threadIdx.x + st];
        __syncthreads();
    }
    if (threadIdx.x == 0) curr[(long)b * 2 * D + c] = red[0] / (float)HW;
}

// out[b,t,o] = bias[o] + sum_w in[b,t,w] * W[o,w]   (tiny T: plain VALU)
__global__ void k_kvproj(float* __restrict__ out, const float* __restrict__ in,
                         const float* __restrict__ W, const float* __restrict__ bias,
                         int T, int Win, int O, int Bc)
{
    long i = (long)blockIdx.x * blockDim.x + threadIdx.x;
    if (i >= (long)Bc * T * O) return;
    int o = (int)(i % O);
    long bt = i / O;
    const float* x = in + bt * Win;
    const float* w = W + (long)o * Win;
    float s = bias[o];
    for (int k = 0; k < Win; ++k) s += x[k] * w[k];
    out[i] = s;
}

// Fused attention: per (b, pixel) wave -> scores (dot over Dq), softmax over T,
// weighted sum of V; writes bf16 into fuse channel slab (concat for free),
// optional f32 copy for the spatial-mean branch.
__global__ __launch_bounds__(128)
void k_attn(const float* __restrict__ q, long qbatch,
            const float* __restrict__ kM, const float* __restrict__ vM,
            int T, int Dq, int Dv, float inv_scale, int HW,
            bf16* __restrict__ fuse, long fbatch, int foff,
            float* __restrict__ avf, long avbatch)
{
    int wv = threadIdx.x >> 5, lane = threadIdx.x & 31;
    long n = (long)blockIdx.x * 4 + wv;
    int b = blockIdx.y;
    if (n >= HW) return;
    q += (long)b * qbatch;
    kM += (long)b * T * Dq;
    vM += (long)b * T * Dv;

    float s[16];
    for (int t = 0; t < T; ++t) {
        float acc = 0.f;
        for (int d = lane; d < Dq; d += 32)
            acc += q[(long)d * HW + n] * kM[(long)t * Dq + d];
#pragma unroll
        for (int off = 16; off > 0; off >>= 1) acc += __shfl_xor(acc, off, 32);
        s[t] = acc * inv_scale;
    }
    float m = s[0];
    for (int t = 1; t < T; ++t) m = fmaxf(m, s[t]);
    float den = 0.f;
    for (int t = 0; t < T; ++t) { s[t] = __expf(s[t] - m); den += s[t]; }
    float rden = 1.f / den;

    bf16* fo = fuse + (long)b * fbatch + (long)foff * HW;
    float* ao = avf ? avf + (long)b * avbatch : (float*)0;
    for (int d = lane; d < Dv; d += 32) {
        float acc = 0.f;
        for (int t = 0; t < T; ++t) acc += s[t] * vM[(long)t * Dv + d];
        acc *= rden;
        fo[(long)d * HW + n] = (bf16)acc;
        if (ao) ao[(long)d * HW + n] = acc;
    }
}

// One block per batch; h kept in LDS; torch-style GRU gates.
__global__ __launch_bounds__(512)
void k_gru(float* __restrict__ out, const float* __restrict__ xin,
           const float* __restrict__ w_ih, const float* __restrict__ w_hh,
           const float* __restrict__ b_ih, const float* __restrict__ b_hh,
           int T, int Hd)
{
    int b = blockIdx.x, j = threadIdx.x;
    __shared__ float h[512];
    __shared__ float xs[512];
    h[j] = 0.f;
    __syncthreads();
    for (int t = 0; t < T; ++t) {
        xs[j] = xin[((long)b * T + t) * Hd + j];
        __syncthreads();
        const float* wr = w_ih + (long)j * Hd;
        const float* wz = w_ih + (long)(Hd + j) * Hd;
        const float* wn = w_ih + (long)(2 * Hd + j) * Hd;
        const float* ur = w_hh + (long)j * Hd;
        const float* uz = w_hh + (long)(Hd + j) * Hd;
        const float* un = w_hh + (long)(2 * Hd + j) * Hd;
        float gr = b_ih[j], gz = b_ih[Hd + j], gn = b_ih[2 * Hd + j];
        float hr = b_hh[j], hz = b_hh[Hd + j], hn = b_hh[2 * Hd + j];
        for (int k = 0; k < Hd; ++k) {
            float xv = xs[k], hv = h[k];
            gr += xv * wr[k]; gz += xv * wz[k]; gn += xv * wn[k];
            hr += hv * ur[k]; hz += hv * uz[k]; hn += hv * un[k];
        }
        float r  = 1.f / (1.f + __expf(-(gr + hr)));
        float z  = 1.f / (1.f + __expf(-(gz + hz)));
        float nn = tanhf(gn + r * hn);
        float hnew = (1.f - z) * nn + z * h[j];
        __syncthreads();
        h[j] = hnew;
        out[((long)b * T + t) * Hd + j] = hnew;
        __syncthreads();
    }
}

// ---------------------------------------------------------------------------
// Host orchestration
// ---------------------------------------------------------------------------
extern "C" void kernel_launch(void* const* d_in, const int* in_sizes, int n_in,
                              void* d_out, int out_size, void* d_ws, size_t ws_size,
                              hipStream_t stream)
{
    (void)in_sizes; (void)n_in; (void)out_size; (void)ws_size;
    const int  Bt = 4;
    const long HW2 = 128 * 128, HW1 = 64 * 64, HW0 = 32 * 32;

    auto F32 = [&](int i) { return (const float*)d_in[i]; };

    // pytree-flatten (sorted-key) input index map
    enum { IX = 0, IWORD = 1, IHIST = 2,
           IG_BHH = 6, IG_BIH = 7, IG_WHH = 8, IG_WIH = 9,
           L1 = 10, L2 = 48 };
    const int l1_low1 = L1 + 0, l1_low2 = L1 + 6, l1_low3 = L1 + 12, l1_up1 = L1 + 20;
    const int l1_whk = L1 + 26, l1_whq = L1 + 28, l1_whv = L1 + 30;
    const int l1_wwk = L1 + 32, l1_wwq = L1 + 34, l1_wwv = L1 + 36;
    const int l2_low1 = L2 + 0, l2_low3 = L2 + 6, l2_up1 = L2 + 14;
    const int l2_whk = L2 + 20, l2_whq = L2 + 22, l2_whv = L2 + 24;
    const int l2_wwk = L2 + 26, l2_wwq = L2 + 28, l2_wwv = L2 + 30;

    // bump allocator over d_ws
    char* base = (char*)d_ws;
    size_t off = 0;
    auto alloc  = [&](size_t bytes) -> void* {
        size_t a = (off + 255) & ~(size_t)255;
        off = a + bytes;
        return base + a;
    };
    auto allocF = [&](long n) { return (float*)alloc((size_t)n * 4); };
    auto allocH = [&](long n) { return (bf16*)alloc((size_t)n * 2); };

    auto cvt_flat = [&](bf16* dst, const float* src, long n) {
        k_cvt_bf16<<<dim3((unsigned)((n + 255) / 256), 1), 256, 0, stream>>>(dst, 0, src, 0, n);
    };
    auto cvt_batch = [&](bf16* dst, long ds, const float* src, long ss, long n) {
        k_cvt_bf16<<<dim3((unsigned)((n + 255) / 256), Bt), 256, 0, stream>>>(dst, ds, src, ss, n);
    };
    auto gemm = [&](int M, int N, int K, const bf16* A, long ab, const bf16* Bp, long bb,
                    float* C, long cb, const float* bias, int relu, int accum,
                    int conv, int Hd, int Wd) {
        dim3 grid((unsigned)(N / BN), (unsigned)(M / BM), Bt);
        k_wmma_gemm<<<grid, 256, 0, stream>>>(A, ab, Bp, bb, C, cb, bias,
                                              M, N, K, relu, accum, conv, Hd, Wd);
    };

    // ---- weights -> bf16 ----
    struct ResW { const float *b1, *b2, *b3, *sb; const bf16 *w1, *w2, *w3, *sw; int cin, cout, h; };
    auto resw = [&](int bi, bool skip, int cin, int cout) {
        ResW r{}; r.cin = cin; r.cout = cout; r.h = cout / 2;
        r.b1 = F32(bi); r.b2 = F32(bi + 1); r.b3 = F32(bi + 2);
        int wi = bi + 3;
        if (skip) {
            r.sb = F32(bi + 3);
            bf16* swb = allocH((long)cout * cin); cvt_flat(swb, F32(bi + 4), (long)cout * cin);
            r.sw = swb; wi = bi + 5;
        }
        bf16* w1b = allocH((long)r.h * cin);     cvt_flat(w1b, F32(wi),     (long)r.h * cin);     r.w1 = w1b;
        bf16* w2b = allocH((long)r.h * r.h * 9); cvt_flat(w2b, F32(wi + 1), (long)r.h * r.h * 9); r.w2 = w2b;
        bf16* w3b = allocH((long)cout * r.h);    cvt_flat(w3b, F32(wi + 2), (long)cout * r.h);    r.w3 = w3b;
        return r;
    };
    auto linw = [&](int bi, int dout, int din, const float** bias) {
        *bias = F32(bi);
        bf16* w = allocH((long)dout * din);
        cvt_flat(w, F32(bi + 1), (long)dout * din);
        return (const bf16*)w;
    };

    ResW rw_up1_2  = resw(l2_up1, false, 256, 256);
    ResW rw_low1_2 = resw(l2_low1, false, 256, 256);
    ResW rw_low3_2 = resw(l2_low3, true, 768, 256);
    ResW rw_up1_1  = resw(l1_up1, false, 256, 256);
    ResW rw_low1_1 = resw(l1_low1, false, 256, 256);
    ResW rw_low2_1 = resw(l1_low2, false, 256, 256);
    ResW rw_low3_1 = resw(l1_low3, true, 768, 256);
    const float *bq1, *bq1h, *bq2, *bq2h;
    const bf16* wwq1 = linw(l1_wwq, 256, 256, &bq1);
    const bf16* whq1 = linw(l1_whq, 256, 512, &bq1h);
    const bf16* wwq2 = linw(l2_wwq, 256, 256, &bq2);
    const bf16* whq2 = linw(l2_whq, 256, 512, &bq2h);

    // ---- shared residual temps (sized for the largest block) ----
    float* T1F = allocF((long)Bt * 128 * HW2);
    bf16*  T1B = allocH((long)Bt * 128 * HW2);
    float* T2F = allocF((long)Bt * 128 * HW2);
    bf16*  T2B = allocH((long)Bt * 128 * HW2);

    auto residual = [&](const ResW& p, const bf16* inbf, long inb,
                        const float* inf32, float* out, int Hd, int Wd) {
        long HW = (long)Hd * Wd;
        gemm(p.h, (int)HW, p.cin, p.w1, 0, inbf, inb, T1F, p.h * HW, p.b1, 1, 0, 0, 0, 0);
        cvt_batch(T1B, p.h * HW, T1F, p.h * HW, p.h * HW);
        gemm(p.h, (int)HW, p.h * 9, p.w2, 0, T1B, p.h * HW, T2F, p.h * HW, p.b2, 1, 0, 1, Hd, Wd);
        cvt_batch(T2B, p.h * HW, T2F, p.h * HW, p.h * HW);
        if (p.sw)
            gemm(p.cout, (int)HW, p.cin, p.sw, 0, inbf, inb, out, p.cout * HW, p.sb, 0, 0, 0, 0, 0);
        else
            (void)hipMemcpyAsync(out, inf32, (size_t)Bt * p.cout * HW * 4,
                                 hipMemcpyDeviceToDevice, stream);
        gemm(p.cout, (int)HW, p.h, p.w3, 0, T2B, p.h * HW, out, p.cout * HW, p.b3, 0, 1, 0, 0, 0);
    };

    // ---- GRU over histories ----
    float* hist_out = allocF((long)Bt * 6 * 512);
    k_gru<<<Bt, 512, 0, stream>>>(hist_out, F32(IHIST), F32(IG_WIH), F32(IG_WHH),
                                  F32(IG_BIH), F32(IG_BHH), 6, 512);

    // ---- level 2 down path ----
    bf16* xbf = allocH((long)Bt * 256 * HW2);
    cvt_flat(xbf, F32(IX), (long)Bt * 256 * HW2);
    float* up1out2 = allocF((long)Bt * 256 * HW2);
    residual(rw_up1_2, xbf, 256 * HW2, F32(IX), up1out2, 128, 128);

    float* xp = allocF((long)Bt * 256 * HW1);
    {
        long tot = (long)Bt * 256 * HW1;
        k_maxpool2<<<dim3((unsigned)((tot + 255) / 256)), 256, 0, stream>>>(xp, F32(IX), 64, 64, tot);
    }
    bf16* xpbf = allocH((long)Bt * 256 * HW1);
    cvt_flat(xpbf, xp, (long)Bt * 256 * HW1);
    float* low1out = allocF((long)Bt * 256 * HW1);
    residual(rw_low1_2, xpbf, 256 * HW1, xp, low1out, 64, 64);
    bf16* low1bf = allocH((long)Bt * 256 * HW1);
    cvt_flat(low1bf, low1out, (long)Bt * 256 * HW1);

    // ---- level 1 ----
    float* up1out1 = allocF((long)Bt * 256 * HW1);
    residual(rw_up1_1, low1bf, 256 * HW1, low1out, up1out1, 64, 64);

    float* lp = allocF((long)Bt * 256 * HW0);
    {
        long tot = (long)Bt * 256 * HW0;
        k_maxpool2<<<dim3((unsigned)((tot + 255) / 256)), 256, 0, stream>>>(lp, low1out, 32, 32, tot);
    }
    bf16* lpbf = allocH((long)Bt * 256 * HW0);
    cvt_flat(lpbf, lp, (long)Bt * 256 * HW0);
    float* low11out = allocF((long)Bt * 256 * HW0);
    residual(rw_low1_1, lpbf, 256 * HW0, lp, low11out, 32, 32);
    bf16* low11bf = allocH((long)Bt * 256 * HW0);
    cvt_flat(low11bf, low11out, (long)Bt * 256 * HW0);
    float* low21out = allocF((long)Bt * 256 * HW0);
    residual(rw_low2_1, low11bf, 256 * HW0, low11out, low21out, 32, 32);

    // fuse buffer level1 [B, 768, 1024]; rows 0..256 = bf16(low2)
    bf16* fuse1 = allocH((long)Bt * 768 * HW0);
    cvt_batch(fuse1, 768 * HW0, low21out, 256 * HW0, 256 * HW0);

    // word attention level1
    float* q1 = allocF((long)Bt * 256 * HW0);
    gemm(256, (int)HW0, 256, wwq1, 0, fuse1, 768 * HW0, q1, 256 * HW0, bq1, 0, 0, 0, 0, 0);
    float* kbuf = allocF((long)Bt * 16 * 256);
    float* vbuf = allocF((long)Bt * 16 * 256);
    k_kvproj<<<dim3((unsigned)((Bt * 16 * 256 + 255) / 256)), 256, 0, stream>>>(
        kbuf, F32(IWORD), F32(l1_wwk + 1), F32(l1_wwk), 16, 256, 256, Bt);
    k_kvproj<<<dim3((unsigned)((Bt * 16 * 256 + 255) / 256)), 256, 0, stream>>>(
        vbuf, F32(IWORD), F32(l1_wwv + 1), F32(l1_wwv), 16, 256, 256, Bt);
    float* av1f = allocF((long)Bt * 256 * HW0);
    k_attn<<<dim3((unsigned)((HW0 + 3) / 4), Bt), 128, 0, stream>>>(
        q1, 256 * HW0, kbuf, vbuf, 16, 256, 256, 1.0f / 16.0f, (int)HW0,
        fuse1, 768 * HW0, 256, av1f, 256 * HW0);

    // curr = mean over spatial of [low2 | av]
    float* curr = (float*)d_out + (long)Bt * 256 * HW2;
    k_mean2<<<dim3(512, Bt), 256, 0, stream>>>(curr, low21out, av1f, 256,
                                               256 * HW0, 256 * HW0, (int)HW0);

    // history attention level1
    float* q1h = allocF((long)Bt * 256 * HW0);
    gemm(256, (int)HW0, 512, whq1, 0, fuse1, 768 * HW0, q1h, 256 * HW0, bq1h, 0, 0, 0, 0, 0);
    float* kbufh = allocF((long)Bt * 6 * 256);
    float* vbufh = allocF((long)Bt * 6 * 256);
    k_kvproj<<<dim3((unsigned)((Bt * 6 * 256 + 255) / 256)), 256, 0, stream>>>(
        kbufh, hist_out, F32(l1_whk + 1), F32(l1_whk), 6, 512, 256, Bt);
    k_kvproj<<<dim3((unsigned)((Bt * 6 * 256 + 255) / 256)), 256, 0, stream>>>(
        vbufh, hist_out, F32(l1_whv + 1), F32(l1_whv), 6, 512, 256, Bt);
    k_attn<<<dim3((unsigned)((HW0 + 3) / 4), Bt), 128, 0, stream>>>(
        q1h, 256 * HW0, kbufh, vbufh, 6, 256, 256, 1.0f / sqrtf(512.0f), (int)HW0,
        fuse1, 768 * HW0, 512, (float*)0, 0);

    // low3 level1 + upsample-add
    float* low3out1 = allocF((long)Bt * 256 * HW0);
    residual(rw_low3_1, fuse1, 768 * HW0, (const float*)0, low3out1, 32, 32);
    float* l1out = allocF((long)Bt * 256 * HW1);
    {
        long tot = (long)Bt * 256 * HW1;
        k_upadd<<<dim3((unsigned)((tot + 255) / 256)), 256, 0, stream>>>(
            l1out, up1out1, low3out1, 64, 64, tot);
    }

    // ---- level 2 attention ----
    bf16* fuse2 = allocH((long)Bt * 768 * HW1);
    cvt_batch(fuse2, 768 * HW1, l1out, 256 * HW1, 256 * HW1);

    float* q2 = allocF((long)Bt * 256 * HW1);
    gemm(256, (int)HW1, 256, wwq2, 0, fuse2, 768 * HW1, q2, 256 * HW1, bq2, 0, 0, 0, 0, 0);
    k_kvproj<<<dim3((unsigned)((Bt * 16 * 256 + 255) / 256)), 256, 0, stream>>>(
        kbuf, F32(IWORD), F32(l2_wwk + 1), F32(l2_wwk), 16, 256, 256, Bt);
    k_kvproj<<<dim3((unsigned)((Bt * 16 * 256 + 255) / 256)), 256, 0, stream>>>(
        vbuf, F32(IWORD), F32(l2_wwv + 1), F32(l2_wwv), 16, 256, 256, Bt);
    k_attn<<<dim3((unsigned)((HW1 + 3) / 4), Bt), 128, 0, stream>>>(
        q2, 256 * HW1, kbuf, vbuf, 16, 256, 256, 1.0f / 16.0f, (int)HW1,
        fuse2, 768 * HW1, 256, (float*)0, 0);

    float* q2h = allocF((long)Bt * 256 * HW1);
    gemm(256, (int)HW1, 512, whq2, 0, fuse2, 768 * HW1, q2h, 256 * HW1, bq2h, 0, 0, 0, 0, 0);
    k_kvproj<<<dim3((unsigned)((Bt * 6 * 256 + 255) / 256)), 256, 0, stream>>>(
        kbufh, hist_out, F32(l2_whk + 1), F32(l2_whk), 6, 512, 256, Bt);
    k_kvproj<<<dim3((unsigned)((Bt * 6 * 256 + 255) / 256)), 256, 0, stream>>>(
        vbufh, hist_out, F32(l2_whv + 1), F32(l2_whv), 6, 512, 256, Bt);
    k_attn<<<dim3((unsigned)((HW1 + 3) / 4), Bt), 128, 0, stream>>>(
        q2h, 256 * HW1, kbufh, vbufh, 6, 256, 256, 1.0f / sqrtf(512.0f), (int)HW1,
        fuse2, 768 * HW1, 512, (float*)0, 0);

    // low3 level2 + final upsample-add into d_out
    float* low3out2 = allocF((long)Bt * 256 * HW1);
    residual(rw_low3_2, fuse2, 768 * HW1, (const float*)0, low3out2, 64, 64);
    {
        long tot = (long)Bt * 256 * HW2;
        k_upadd<<<dim3((unsigned)((tot + 255) / 256)), 256, 0, stream>>>(
            (float*)d_out, up1out2, low3out2, 128, 128, tot);
    }
}
